// Decoder_74844100100875
// MI455X (gfx1250) — compile-verified, matrix operand order checked by reading, not run on hardware
//
#include <hip/hip_runtime.h>
#include <math.h>

// ---------------------------------------------------------------------------
// Problem constants (match reference)
// ---------------------------------------------------------------------------
#define BB   128     // batch
#define HH   512     // hidden
#define AA   128     // attention dim
#define TT   256     // encoder time steps
#define CC   256     // vocab
#define H4   2048    // 4*H

typedef __attribute__((ext_vector_type(16))) __bf16 v16bf;
typedef __attribute__((ext_vector_type(8)))  __bf16 v8bf;
typedef __attribute__((ext_vector_type(8)))  float  v8f;

// ---------------------------------------------------------------------------
// WMMA fragment load: row-major bf16 matrix row pointer, 16x32 fragment.
// Per ISA 16-bit A/B layout: lane L -> row/col = L&15; kh = (L>>4)*8;
// elements j0..7 -> k = kbase+kh+j ; j8..15 -> k = kbase+16+kh+j.
// Both halves are contiguous 16-byte chunks -> b128 loads.
// ---------------------------------------------------------------------------
__device__ inline v16bf load_frag16x32(const __bf16* __restrict__ rowptr, int kh) {
    v8bf lo = *reinterpret_cast<const v8bf*>(rowptr + kh);
    v8bf hi = *reinterpret_cast<const v8bf*>(rowptr + 16 + kh);
    v16bf r;
#pragma unroll
    for (int j = 0; j < 8; ++j) { r[j] = lo[j]; r[8 + j] = hi[j]; }
    return r;
}

__device__ inline void gemm_accum(v8f& acc, const __bf16* __restrict__ arow,
                                  const __bf16* __restrict__ brow, int K, int kh) {
    for (int kb = 0; kb < K; kb += 32) {
        v16bf a = load_frag16x32(arow + kb, kh);
        v16bf b = load_frag16x32(brow + kb, kh);
        acc = __builtin_amdgcn_wmma_f32_16x16x32_bf16(false, a, false, b,
                                                      (short)0, acc, false, false);
    }
}

// ---------------------------------------------------------------------------
// Generalized fused GEMM:
//   Out[M,N] = sum_p Ap[M,Kp] @ Wp[N,Kp]^T   (+ bias[N]) (opt LeakyReLU)
// Independent leading dimensions per operand (allows K-split views of a
// wider weight matrix). Optional row-gather on A0 (embedding lookup).
// One wave per 16x16 output tile; f32 accumulation via WMMA bf16.
// ---------------------------------------------------------------------------
__global__ void __launch_bounds__(256)
wmma_gemm3_kernel(const __bf16* __restrict__ A0, int lda0,
                  const __bf16* __restrict__ W0, int ldw0, int K0,
                  const __bf16* __restrict__ A1, int lda1,
                  const __bf16* __restrict__ W1, int ldw1, int K1,
                  const __bf16* __restrict__ A2, int lda2,
                  const __bf16* __restrict__ W2, int ldw2, int K2,
                  const int* __restrict__ gatherY, int gt, int gml,
                  const float* __restrict__ bias, int leaky,
                  float* __restrict__ Out, int Ntiles, int N)
{
    const int wave = (blockIdx.x * blockDim.x + threadIdx.x) >> 5;
    const int lane = threadIdx.x & 31;
    const int mt = wave / Ntiles;
    const int nt = wave % Ntiles;
    const int r  = lane & 15;
    const int kh = (lane >> 4) << 3;

    const int m = mt * 16 + r;            // A row for this lane
    const int n = nt * 16 + r;            // W row for this lane

    const int arow0idx = gatherY ? gatherY[m * gml + gt] : m;

    v8f acc = {};
    gemm_accum(acc, A0 + (size_t)arow0idx * lda0, W0 + (size_t)n * ldw0, K0, kh);
    if (A1 != nullptr)
        gemm_accum(acc, A1 + (size_t)m * lda1, W1 + (size_t)n * ldw1, K1, kh);
    if (A2 != nullptr)
        gemm_accum(acc, A2 + (size_t)m * lda2, W2 + (size_t)n * ldw2, K2, kh);

    // C/D layout: VGPR v -> row = v + 8*(lane>=16); col = lane&15
    const int col   = nt * 16 + r;
    const int rbase = mt * 16 + ((lane >> 4) << 3);
    const float bv  = bias ? bias[col] : 0.0f;
    float* o = Out + (size_t)rbase * N + col;
#pragma unroll
    for (int v = 0; v < 8; ++v) {
        float x = acc[v] + bv;
        if (leaky) x = (x >= 0.0f) ? x : 0.01f * x;
        o[(size_t)v * N] = x;
    }
}

// ---------------------------------------------------------------------------
// Small helper kernels
// ---------------------------------------------------------------------------
__global__ void f32_to_bf16_kernel(const float* __restrict__ s, __bf16* __restrict__ d, int n) {
    int i = blockIdx.x * blockDim.x + threadIdx.x;
    if (i < n) d[i] = (__bf16)s[i];
}

__global__ void add_vec_kernel(const float* __restrict__ a, const float* __restrict__ b,
                               float* __restrict__ o, int n) {
    int i = blockIdx.x * blockDim.x + threadIdx.x;
    if (i < n) o[i] = a[i] + b[i];
}

__global__ void bcast_state_kernel(const float* __restrict__ src /*[H]*/,
                                   float* __restrict__ dstf, __bf16* __restrict__ dstb) {
    int i = blockIdx.x * blockDim.x + threadIdx.x;   // B*H
    if (i >= BB * HH) return;
    float v = src[i & (HH - 1)];
    dstf[i] = v;
    if (dstb) dstb[i] = (__bf16)v;
}

// Gates already contain bih+bhh (fused into the GEMM epilogue).
__global__ void lstm_pw_kernel(const float* __restrict__ gates /*[B,4H]*/,
                               float* __restrict__ c, float* __restrict__ hf,
                               __bf16* __restrict__ hb) {
    int i = blockIdx.x * blockDim.x + threadIdx.x;   // B*H
    if (i >= BB * HH) return;
    int b = i >> 9, h = i & (HH - 1);
    const float* g = gates + (size_t)b * H4;
    float I = 1.0f / (1.0f + __expf(-g[h]));
    float F = 1.0f / (1.0f + __expf(-g[HH + h]));
    float G = tanhf(g[2 * HH + h]);
    float O = 1.0f / (1.0f + __expf(-g[3 * HH + h]));
    float cn = F * c[i] + I * G;
    float hn = O * tanhf(cn);
    c[i] = cn; hf[i] = hn; hb[i] = (__bf16)hn;
}

// ---------------------------------------------------------------------------
// Attention: one block per batch row (256 threads = T). f32; masked softmax
// (softmax -> mask -> renorm == softmax over masked subset).
// Writes ctx in f32 (for accuracy downstream) and bf16 (for WMMA operands).
// ---------------------------------------------------------------------------
__global__ void __launch_bounds__(256)
attention_kernel(const float* __restrict__ key   /*[B,A,T]*/,
                 const float* __restrict__ value /*[B,T,A]*/,
                 const float* __restrict__ h3    /*[B,H]*/,
                 const float* __restrict__ Wq    /*[A,H]*/,
                 const float* __restrict__ bq,
                 const int*   __restrict__ frame_lens,
                 float* __restrict__ ctx /*[B,A]*/,
                 __bf16* __restrict__ ctxb /*[B,A]*/)
{
    __shared__ float q[AA];
    __shared__ float attn[TT];
    __shared__ float red[TT];
    const int b = blockIdx.x, tid = threadIdx.x;

    if (tid < AA) {
        float s = bq[tid];
        const float* hrow = h3 + (size_t)b * HH;
        const float* wrow = Wq + (size_t)tid * HH;
        for (int k = 0; k < HH; ++k) s += hrow[k] * wrow[k];
        q[tid] = s;
    }
    __syncthreads();

    // energy[t]
    float en = 0.0f;
    {
        const float* kcol = key + (size_t)b * AA * TT + tid;   // key[b,a,t]
        for (int a = 0; a < AA; ++a) en += q[a] * kcol[(size_t)a * TT];
    }
    const int flen = frame_lens[b] / 8;
    const bool valid = tid < flen;

    red[tid] = valid ? en : -3.0e38f;
    __syncthreads();
    for (int s = TT / 2; s > 0; s >>= 1) {
        if (tid < s) red[tid] = fmaxf(red[tid], red[tid + s]);
        __syncthreads();
    }
    const float mx = red[0];
    __syncthreads();

    float ex = valid ? __expf(en - mx) : 0.0f;
    red[tid] = ex;
    __syncthreads();
    for (int s = TT / 2; s > 0; s >>= 1) {
        if (tid < s) red[tid] += red[tid + s];
        __syncthreads();
    }
    attn[tid] = ex / red[0];
    __syncthreads();

    if (tid < AA) {
        float s = 0.0f;
        const float* vp = value + (size_t)b * TT * AA + tid;   // value[b,t,v]
        for (int t = 0; t < TT; ++t) s += attn[t] * vp[(size_t)t * AA];
        ctx[b * AA + tid]  = s;
        ctxb[b * AA + tid] = (__bf16)s;
    }
}

// ---------------------------------------------------------------------------
// Tied projection + log_softmax: one block per batch row (256 threads = C).
// logits = m @ emb^T + bproj; out[b, t, :] = log_softmax(logits)
// ---------------------------------------------------------------------------
__global__ void __launch_bounds__(256)
proj_logsoftmax_kernel(const float* __restrict__ m   /*[B,H]*/,
                       const float* __restrict__ emb /*[C,H] f32*/,
                       const float* __restrict__ bproj,
                       float* __restrict__ out, int t, int maxlen)
{
    __shared__ float mrow[HH];
    __shared__ float red[CC];
    const int b = blockIdx.x, c = threadIdx.x;

    for (int i = c; i < HH; i += blockDim.x) mrow[i] = m[(size_t)b * HH + i];
    __syncthreads();

    float s = bproj[c];
    const float* erow = emb + (size_t)c * HH;
    for (int k = 0; k < HH; ++k) s += mrow[k] * erow[k];

    red[c] = s;
    __syncthreads();
    for (int w = CC / 2; w > 0; w >>= 1) {
        if (c < w) red[c] = fmaxf(red[c], red[c + w]);
        __syncthreads();
    }
    const float mx = red[0];
    __syncthreads();
    float ex = __expf(s - mx);
    red[c] = ex;
    __syncthreads();
    for (int w = CC / 2; w > 0; w >>= 1) {
        if (c < w) red[c] += red[c + w];
        __syncthreads();
    }
    const float lse = mx + __logf(red[0]);
    out[((size_t)b * maxlen + t) * CC + c] = s - lse;
}

// ---------------------------------------------------------------------------
// Host driver
// ---------------------------------------------------------------------------
extern "C" void kernel_launch(void* const* d_in, const int* in_sizes, int n_in,
                              void* d_out, int out_size, void* d_ws, size_t ws_size,
                              hipStream_t stream)
{
    const float* key        = (const float*)d_in[0];
    const float* value      = (const float*)d_in[1];
    const int*   Yinput     = (const int*)  d_in[2];
    const int*   frame_lens = (const int*)  d_in[3];
    const float* emb        = (const float*)d_in[5];
    const float* Wq         = (const float*)d_in[6];
    const float* bq         = (const float*)d_in[7];
    const float* Wih1 = (const float*)d_in[8];  const float* Whh1 = (const float*)d_in[9];
    const float* bih1 = (const float*)d_in[10]; const float* bhh1 = (const float*)d_in[11];
    const float* Wih2 = (const float*)d_in[12]; const float* Whh2 = (const float*)d_in[13];
    const float* bih2 = (const float*)d_in[14]; const float* bhh2 = (const float*)d_in[15];
    const float* Wih3 = (const float*)d_in[16]; const float* Whh3 = (const float*)d_in[17];
    const float* bih3 = (const float*)d_in[18]; const float* bhh3 = (const float*)d_in[19];
    const float* Wmlp = (const float*)d_in[20]; const float* bmlp = (const float*)d_in[21];
    const float* bproj = (const float*)d_in[22];
    const float* h00 = (const float*)d_in[23];
    const float* h01 = (const float*)d_in[24];
    const float* h02 = (const float*)d_in[25];
    const float* c00 = (const float*)d_in[26];
    const float* c01 = (const float*)d_in[27];
    const float* c02 = (const float*)d_in[28];

    const int maxlen = in_sizes[2] / BB;     // 300
    float* out = (float*)d_out;

    // workspace allocator (256B aligned)
    char* w = (char*)d_ws;
    size_t off = 0;
    auto alloc = [&](size_t bytes) -> void* {
        void* p = (void*)(w + off);
        off += (bytes + 255) & ~(size_t)255;
        return p;
    };
    __bf16* wih1b = (__bf16*)alloc((size_t)H4 * (HH + AA) * 2);
    __bf16* whh1b = (__bf16*)alloc((size_t)H4 * HH * 2);
    __bf16* wih2b = (__bf16*)alloc((size_t)H4 * HH * 2);
    __bf16* whh2b = (__bf16*)alloc((size_t)H4 * HH * 2);
    __bf16* wih3b = (__bf16*)alloc((size_t)H4 * HH * 2);
    __bf16* whh3b = (__bf16*)alloc((size_t)H4 * HH * 2);
    __bf16* wmlpb = (__bf16*)alloc((size_t)HH * (HH + AA) * 2);
    __bf16* embb  = (__bf16*)alloc((size_t)CC * HH * 2);
    float*  bsum1 = (float*)alloc((size_t)H4 * 4);
    float*  bsum2 = (float*)alloc((size_t)H4 * 4);
    float*  bsum3 = (float*)alloc((size_t)H4 * 4);
    float*  h1f = (float*)alloc((size_t)BB * HH * 4);
    float*  h2f = (float*)alloc((size_t)BB * HH * 4);
    float*  h3f = (float*)alloc((size_t)BB * HH * 4);
    float*  c1  = (float*)alloc((size_t)BB * HH * 4);
    float*  c2  = (float*)alloc((size_t)BB * HH * 4);
    float*  c3  = (float*)alloc((size_t)BB * HH * 4);
    __bf16* h1b = (__bf16*)alloc((size_t)BB * HH * 2);
    __bf16* h2b = (__bf16*)alloc((size_t)BB * HH * 2);
    __bf16* h3b = (__bf16*)alloc((size_t)BB * HH * 2);
    float*  ctx   = (float*)alloc((size_t)BB * AA * 4);
    __bf16* ctxb  = (__bf16*)alloc((size_t)BB * AA * 2);
    float*  gates = (float*)alloc((size_t)BB * H4 * 4);
    float*  mbuf  = (float*)alloc((size_t)BB * HH * 4);

    auto cvt = [&](const float* s, __bf16* d, int n) {
        f32_to_bf16_kernel<<<(n + 255) / 256, 256, 0, stream>>>(s, d, n);
    };
    // --- one-time weight conversion (f32 -> bf16, WMMA operands) ---
    cvt(Wih1, wih1b, H4 * (HH + AA));
    cvt(Whh1, whh1b, H4 * HH);
    cvt(Wih2, wih2b, H4 * HH);
    cvt(Whh2, whh2b, H4 * HH);
    cvt(Wih3, wih3b, H4 * HH);
    cvt(Whh3, whh3b, H4 * HH);
    cvt(Wmlp, wmlpb, HH * (HH + AA));
    cvt(emb,  embb,  CC * HH);

    // combined LSTM biases (bih + bhh), fused into GEMM epilogue
    add_vec_kernel<<<H4 / 256, 256, 0, stream>>>(bih1, bhh1, bsum1, H4);
    add_vec_kernel<<<H4 / 256, 256, 0, stream>>>(bih2, bhh2, bsum2, H4);
    add_vec_kernel<<<H4 / 256, 256, 0, stream>>>(bih3, bhh3, bsum3, H4);

    // --- initial states ---
    const int gBH = (BB * HH) / 256;
    bcast_state_kernel<<<gBH, 256, 0, stream>>>(h00, h1f, h1b);
    bcast_state_kernel<<<gBH, 256, 0, stream>>>(h01, h2f, h2b);
    bcast_state_kernel<<<gBH, 256, 0, stream>>>(h02, h3f, h3b);
    bcast_state_kernel<<<gBH, 256, 0, stream>>>(c00, c1, nullptr);
    bcast_state_kernel<<<gBH, 256, 0, stream>>>(c01, c2, nullptr);
    bcast_state_kernel<<<gBH, 256, 0, stream>>>(c02, c3, nullptr);

    // initial context from broadcast h3
    attention_kernel<<<BB, TT, 0, stream>>>(key, value, h3f, Wq, bq, frame_lens, ctx, ctxb);

    const int gemmLstmBlocks = (BB / 16) * (H4 / 16) * 32 / 256;  // 128
    const int gemmMlpBlocks  = (BB / 16) * (HH / 16) * 32 / 256;  // 32

    for (int t = 0; t < maxlen; ++t) {
        // LSTM 1: gates = emb[y_t]@Wih1[:, :H]^T + ctx@Wih1[:, H:]^T + h1@Whh1^T + bias
        wmma_gemm3_kernel<<<gemmLstmBlocks, 256, 0, stream>>>(
            embb, HH,      wih1b,        HH + AA, HH,
            ctxb, AA,      wih1b + HH,   HH + AA, AA,
            h1b,  HH,      whh1b,        HH,      HH,
            Yinput, t, maxlen, bsum1, 0, gates, H4 / 16, H4);
        lstm_pw_kernel<<<gBH, 256, 0, stream>>>(gates, c1, h1f, h1b);

        // LSTM 2: gates = h1@Wih2^T + h2@Whh2^T + bias
        wmma_gemm3_kernel<<<gemmLstmBlocks, 256, 0, stream>>>(
            h1b, HH, wih2b, HH, HH,
            h2b, HH, whh2b, HH, HH,
            nullptr, 0, nullptr, 0, 0,
            nullptr, 0, 0, bsum2, 0, gates, H4 / 16, H4);
        lstm_pw_kernel<<<gBH, 256, 0, stream>>>(gates, c2, h2f, h2b);

        // LSTM 3
        wmma_gemm3_kernel<<<gemmLstmBlocks, 256, 0, stream>>>(
            h2b, HH, wih3b, HH, HH,
            h3b, HH, whh3b, HH, HH,
            nullptr, 0, nullptr, 0, 0,
            nullptr, 0, 0, bsum3, 0, gates, H4 / 16, H4);
        lstm_pw_kernel<<<gBH, 256, 0, stream>>>(gates, c3, h3f, h3b);

        // attention -> ctx (f32 + bf16)
        attention_kernel<<<BB, TT, 0, stream>>>(key, value, h3f, Wq, bq, frame_lens, ctx, ctxb);

        // MLP: m = LeakyReLU(h3@Wmlp[:, :H]^T + ctx@Wmlp[:, H:]^T + bmlp)
        wmma_gemm3_kernel<<<gemmMlpBlocks, 256, 0, stream>>>(
            h3b,  HH, wmlpb,      HH + AA, HH,
            ctxb, AA, wmlpb + HH, HH + AA, AA,
            nullptr, 0, nullptr, 0, 0,
            nullptr, 0, 0, bmlp, 1, mbuf, HH / 16, HH);

        // tied projection + log_softmax -> out[:, t, :]
        proj_logsoftmax_kernel<<<BB, CC, 0, stream>>>(mbuf, emb, bproj, out, t, maxlen);
    }
}
